// DoubleStreamBlock_16904991277495
// MI455X (gfx1250) — matrix operand order repaired and checked by reading, not compiled, not run on hardware
//
#include <hip/hip_runtime.h>
#include <math.h>

// ---------------- types / helpers ----------------
typedef __attribute__((ext_vector_type(16))) __bf16 v16bf;
typedef __attribute__((ext_vector_type(8)))  float  v8f;

union V16U { v16bf v; unsigned short u[16]; };

__device__ __forceinline__ unsigned short f2bu(float f){
  unsigned int x = __float_as_uint(f);
  x += 0x7fffu + ((x >> 16) & 1u);           // RNE
  return (unsigned short)(x >> 16);
}
__device__ __forceinline__ float bu2f(unsigned short u){
  return __uint_as_float(((unsigned int)u) << 16);
}
__device__ __forceinline__ float gelu_tanh(float x){
  float x3 = x*x*x;
  return 0.5f * x * (1.f + tanhf(0.7978845608028654f * (x + 0.044715f * x3)));
}
__device__ __forceinline__ v8f wmma_bf16(const V16U& a, const V16U& b, v8f c){
  return __builtin_amdgcn_wmma_f32_16x16x32_bf16(false, a.v, false, b.v, (short)0, c, false, false);
}

// async global -> LDS copy (16 bytes per lane), tracked by ASYNCcnt
__device__ __forceinline__ void async_copy_b128(unsigned lds_off, const void* gptr){
  asm volatile("global_load_async_to_lds_b128 %0, %1, off"
               :: "v"(lds_off), "v"(gptr) : "memory");
}
__device__ __forceinline__ void wait_async0(){
#if __has_builtin(__builtin_amdgcn_s_wait_asynccnt)
  __builtin_amdgcn_s_wait_asynccnt(0);
#else
  asm volatile("s_wait_asynccnt 0" ::: "memory");
#endif
}
// low 32 bits of a flat shared-memory pointer = LDS byte address
__device__ __forceinline__ unsigned lds_addr(const void* p){
  return (unsigned)(unsigned long long)p;
}

// ---------------- fused LayerNorm + modulation -> bf16 ----------------
__global__ __launch_bounds__(256) void modln_kernel(
    const float* __restrict__ x, const float* __restrict__ shift,
    const float* __restrict__ scale, unsigned short* __restrict__ out, int Mb)
{
  const int HIDc = 3072;
  int row = blockIdx.x;
  int b = row / Mb;
  const float* xr = x + (size_t)row * HIDc;
  float s = 0.f, s2 = 0.f;
  for (int i = threadIdx.x; i < HIDc; i += 256){ float v = xr[i]; s += v; s2 += v*v; }
  #pragma unroll
  for (int m = 16; m > 0; m >>= 1){ s += __shfl_xor(s, m, 32); s2 += __shfl_xor(s2, m, 32); }
  __shared__ float red[2][8];
  int wid = threadIdx.x >> 5, lane = threadIdx.x & 31;
  if (lane == 0){ red[0][wid] = s; red[1][wid] = s2; }
  __syncthreads();
  float ts = 0.f, ts2 = 0.f;
  #pragma unroll
  for (int i = 0; i < 8; i++){ ts += red[0][i]; ts2 += red[1][i]; }
  float mean = ts * (1.f/3072.f);
  float var  = ts2 * (1.f/3072.f) - mean*mean;
  float rinv = rsqrtf(var + 1e-6f);
  const float* sh = shift + (size_t)b * HIDc;
  const float* sc = scale + (size_t)b * HIDc;
  unsigned short* o = out + (size_t)row * HIDc;
  for (int i = threadIdx.x; i < HIDc; i += 256){
    float v = (xr[i] - mean) * rinv;
    o[i] = f2bu((1.f + sc[i]) * v + sh[i]);
  }
}

// ---------------- WMMA GEMM: C[M x N] = A(bf16)[M x K] @ W(f32)[N x K]^T + bias ----------------
// 64x64 block tile, BK=32, double-buffered LDS, async A staging, 1 barrier/K-step.
// EPI 0: Cf = acc+bias (fp32)
// EPI 1: Cb = bf16(gelu(acc+bias))
// EPI 2: Cf = res + gate[b,n] * (acc+bias)
template<int EPI>
__global__ __launch_bounds__(256) void gemm_kernel(
    const unsigned short* __restrict__ A, const float* __restrict__ W,
    const float* __restrict__ bias, const float* __restrict__ res,
    const float* __restrict__ gate, float* __restrict__ Cf,
    unsigned short* __restrict__ Cb,
    int Mb, int N, int K, int A_batch_rows, int A_row_off)
{
  constexpr int AS = 40;  // LDS A row stride (halfs)
  constexpr int BS = 68;  // LDS B^T row stride (halfs); Bs[k][n]
  __shared__ unsigned short As[2][64 * AS];
  __shared__ unsigned short Bs[2][32 * BS];

  int tm = blockIdx.x, tn = blockIdx.y;
  int w = threadIdx.x >> 5, lane = threadIdx.x & 31;
  int wm = w & 3, wn = w >> 2;            // 4 x 2 waves -> 64 x 64 tile
  int half = lane >> 4, l16 = lane & 15;

  v8f c0, c1;
  #pragma unroll
  for (int i = 0; i < 8; i++){ c0[i] = 0.f; c1[i] = 0.f; }

  // staging roles (shared by A and W paths)
  int r64 = threadIdx.x >> 2;             // 0..63
  int c8  = (threadIdx.x & 3) * 8;        // 0,8,16,24
  int gm_s = tm*64 + r64;
  int bA = gm_s / Mb, rA = gm_s % Mb;
  const unsigned short* Arow = A + ((size_t)bA * A_batch_rows + A_row_off + rA) * K + c8;
  const float* Wrow = W + ((size_t)(tn*64 + r64)) * K + c8;
  unsigned as_off[2] = { lds_addr(&As[0][r64*AS + c8]), lds_addr(&As[1][r64*AS + c8]) };

  int T = K >> 5;
  // prologue: stage tile 0
  async_copy_b128(as_off[0], Arow);
  float4 wa = *(const float4*)(Wrow);
  float4 wb = *(const float4*)(Wrow + 4);

  for (int t = 0; t < T; t++){
    int cur = t & 1, nxt = cur ^ 1;
    // convert + store W regs into Bs[cur] (k-major transposed)
    {
      float f[8] = {wa.x, wa.y, wa.z, wa.w, wb.x, wb.y, wb.z, wb.w};
      #pragma unroll
      for (int i = 0; i < 8; i++) Bs[cur][(c8 + i)*BS + r64] = f2bu(f[i]);
    }
    wait_async0();          // A(t) landed in As[cur]
    __syncthreads();        // tile t fully staged, everyone done with tile t-1
    if (t + 1 < T){         // prefetch tile t+1 (overlaps compute below)
      async_copy_b128(as_off[nxt], Arow + (t + 1) * 32);
      wa = *(const float4*)(Wrow + (t + 1) * 32);
      wb = *(const float4*)(Wrow + (t + 1) * 32 + 4);
    }
    // compute on tile t
    V16U a, b0, b1;
    const unsigned short* Ap = &As[cur][(wm*16 + l16)*AS];
    #pragma unroll
    for (int i = 0; i < 8; i++){
      int kk = (i>>2)*16 + half*8 + (i&3)*2;
      a.u[2*i]   = Ap[kk];
      a.u[2*i+1] = Ap[kk+1];
    }
    const unsigned short* Bp = &Bs[cur][wn*32 + l16];
    #pragma unroll
    for (int j = 0; j < 16; j++){ int kb = (j>>3)*16 + half*8 + (j&7); b0.u[j] = Bp[kb*BS]; b1.u[j] = Bp[kb*BS + 16]; }
    c0 = wmma_bf16(a, b0, c0);
    c1 = wmma_bf16(a, b1, c1);
  }

  int col0 = tn*64 + wn*32 + l16;
  int col1 = col0 + 16;
  float bv0 = bias[col0], bv1 = bias[col1];
  #pragma unroll
  for (int r = 0; r < 8; r++){
    int grow = tm*64 + wm*16 + half*8 + r;
    float v0 = c0[r] + bv0;
    float v1 = c1[r] + bv1;
    size_t o0 = (size_t)grow * N + col0;
    size_t o1 = (size_t)grow * N + col1;
    if (EPI == 0){
      Cf[o0] = v0; Cf[o1] = v1;
    } else if (EPI == 1){
      Cb[o0] = f2bu(gelu_tanh(v0)); Cb[o1] = f2bu(gelu_tanh(v1));
    } else {
      int bb = grow / Mb;
      float g0 = gate[(size_t)bb * N + col0];
      float g1 = gate[(size_t)bb * N + col1];
      Cf[o0] = res[o0] + g0 * v0;
      Cf[o1] = res[o1] + g1 * v1;
    }
  }
}

// ---------------- QKV post: RMS-norm + RoPE, scatter to [B,NH,L,HD] bf16 ----------------
__global__ __launch_bounds__(128) void qkv_post_kernel(
    const float* __restrict__ QKVimg, const float* __restrict__ QKVtxt,
    const float* __restrict__ pe,
    const float* __restrict__ qsi, const float* __restrict__ ksi,
    const float* __restrict__ qst, const float* __restrict__ kst,
    unsigned short* __restrict__ Q, unsigned short* __restrict__ Kb,
    unsigned short* __restrict__ V)
{
  constexpr int L = 1280, HD = 128, NH = 24;
  int idx = blockIdx.x, d = threadIdx.x;
  int tok = idx % L;
  int h   = (idx / L) % NH;
  int b   = idx / (L * NH);
  bool isimg = tok < 1024;
  const float* src = isimg ? (QKVimg + ((size_t)(b*1024 + tok)) * 9216)
                           : (QKVtxt + ((size_t)(b*256 + tok - 1024)) * 9216);
  float qv = src[h*HD + d];
  float kv = src[3072 + h*HD + d];
  float vv = src[6144 + h*HD + d];
  float sq = qv*qv, sk = kv*kv;
  #pragma unroll
  for (int m = 16; m > 0; m >>= 1){ sq += __shfl_xor(sq, m, 32); sk += __shfl_xor(sk, m, 32); }
  __shared__ float part[2][4];
  int wid = d >> 5, lane = d & 31;
  if (lane == 0){ part[0][wid] = sq; part[1][wid] = sk; }
  __syncthreads();
  sq = part[0][0]+part[0][1]+part[0][2]+part[0][3];
  sk = part[1][0]+part[1][1]+part[1][2]+part[1][3];
  float qr = rsqrtf(sq * (1.f/128.f) + 1e-6f) * (isimg ? qsi[d] : qst[d]);
  float kr = rsqrtf(sk * (1.f/128.f) + 1e-6f) * (isimg ? ksi[d] : kst[d]);
  __shared__ float qn[128], kn[128];
  qn[d] = qv * qr; kn[d] = kv * kr;
  __syncthreads();
  int p = d >> 1, j = d & 1;
  const float* per = pe + ((((size_t)b * L + tok) * 64 + p) * 2 + j) * 2;
  float qo = per[0]*qn[2*p] + per[1]*qn[2*p+1];
  float ko = per[0]*kn[2*p] + per[1]*kn[2*p+1];
  size_t o = (((size_t)(b*NH + h)) * L + tok) * HD + d;
  Q[o] = f2bu(qo); Kb[o] = f2bu(ko); V[o] = f2bu(vv);
}

// ---------------- flash attention (wave32 WMMA, async V staging) ----------------
__global__ __launch_bounds__(256) void attn_kernel(
    const unsigned short* __restrict__ Qb, const unsigned short* __restrict__ Kb,
    const unsigned short* __restrict__ Vb, const int* __restrict__ tseq,
    unsigned short* __restrict__ out)
{
  constexpr int L = 1280, HD = 128, NH = 24, HIDc = 3072;
  __shared__ unsigned short Kt[128][36];     // [d][key]  (transposed K tile)
  __shared__ unsigned short Vt[32][136];     // [key][d]
  __shared__ unsigned short Pt[8][16*36];    // per-wave P scratch

  int blk = blockIdx.x;
  int qt = blk % 10;
  int h  = (blk / 10) % NH;
  int b  = blk / (10 * NH);
  int w = threadIdx.x >> 5, lane = threadIdx.x & 31;
  int half = lane >> 4, l16 = lane & 15;
  int kvalid = 1024 + tseq[b];
  size_t bh = (size_t)(b * NH + h);

  // preload q fragments, pre-scaled by 1/sqrt(HD)
  V16U qa[4];
  {
    int qrow = qt*128 + w*16 + l16;
    const unsigned short* qp = Qb + (bh * L + qrow) * HD;
    #pragma unroll
    for (int c = 0; c < 4; c++){
      #pragma unroll
      for (int i = 0; i < 8; i++){
        int kk = c*32 + (i>>2)*16 + half*8 + (i&3)*2;
        qa[c].u[2*i]   = f2bu(bu2f(qp[kk])   * 0.08838834764831845f);
        qa[c].u[2*i+1] = f2bu(bu2f(qp[kk+1]) * 0.08838834764831845f);
      }
    }
  }

  float mrow[8], lrow[8];
  v8f acc[8];
  #pragma unroll
  for (int r = 0; r < 8; r++){ mrow[r] = -1e38f; lrow[r] = 0.f; }
  #pragma unroll
  for (int n = 0; n < 8; n++)
    #pragma unroll
    for (int r = 0; r < 8; r++) acc[n][r] = 0.f;

  int skey = threadIdx.x >> 3;          // 0..31
  int sdg  = (threadIdx.x & 7) * 16;    // 0..112
  unsigned vt_off = lds_addr(&Vt[skey][sdg]);

  for (int kt = 0; kt < L/32; kt++){
    __syncthreads();   // everyone done reading previous Kt/Vt
    {
      const unsigned short* vsrc = Vb + (bh*L + kt*32 + skey)*HD + sdg;
      async_copy_b128(vt_off,      vsrc);
      async_copy_b128(vt_off + 16, vsrc + 8);
      const uint4* kr4 = (const uint4*)(Kb + (bh*L + kt*32 + skey)*HD + sdg);
      uint4 a0 = kr4[0], a1 = kr4[1];
      unsigned int t0[8] = {a0.x,a0.y,a0.z,a0.w,a1.x,a1.y,a1.z,a1.w};
      #pragma unroll
      for (int i = 0; i < 8; i++){
        Kt[sdg + 2*i][skey]     = (unsigned short)(t0[i] & 0xffffu);
        Kt[sdg + 2*i + 1][skey] = (unsigned short)(t0[i] >> 16);
      }
      wait_async0();
    }
    __syncthreads();

    // S = q @ K^T  (two 16x16 key subtiles, K-dim 128 in 4 chunks)
    v8f s0, s1;
    #pragma unroll
    for (int r = 0; r < 8; r++){ s0[r] = 0.f; s1[r] = 0.f; }
    #pragma unroll
    for (int c = 0; c < 4; c++){
      V16U b0, b1;
      #pragma unroll
      for (int j = 0; j < 16; j++){ int kd = c*32 + (j>>3)*16 + half*8 + (j&7); b0.u[j] = Kt[kd][l16]; b1.u[j] = Kt[kd][16 + l16]; }
      s0 = wmma_bf16(qa[c], b0, s0);
      s1 = wmma_bf16(qa[c], b1, s1);
    }

    int key0 = kt*32 + l16, key1 = key0 + 16;
    bool ok0 = key0 < kvalid, ok1 = key1 < kvalid;
    #pragma unroll
    for (int r = 0; r < 8; r++){
      float f0 = ok0 ? s0[r] : -1e30f;
      float f1 = ok1 ? s1[r] : -1e30f;
      float x = fmaxf(f0, f1);
      #pragma unroll
      for (int m = 8; m > 0; m >>= 1) x = fmaxf(x, __shfl_xor(x, m, 32));
      float nm = fmaxf(mrow[r], x);
      float p0 = __expf(f0 - nm), p1 = __expf(f1 - nm);
      float sm = p0 + p1;
      #pragma unroll
      for (int m = 8; m > 0; m >>= 1) sm += __shfl_xor(sm, m, 32);
      float esc = __expf(mrow[r] - nm);
      lrow[r] = lrow[r] * esc + sm;
      mrow[r] = nm;
      #pragma unroll
      for (int n = 0; n < 8; n++) acc[n][r] *= esc;
      Pt[w][(half*8 + r)*36 + l16]      = f2bu(p0);
      Pt[w][(half*8 + r)*36 + 16 + l16] = f2bu(p1);
    }

    // reload P as A-fragment, accumulate O += P @ V
    V16U pa;
    #pragma unroll
    for (int i = 0; i < 8; i++){
      int kk = (i>>2)*16 + half*8 + (i&3)*2;
      pa.u[2*i]   = Pt[w][l16*36 + kk];
      pa.u[2*i+1] = Pt[w][l16*36 + kk + 1];
    }
    #pragma unroll
    for (int n = 0; n < 8; n++){
      V16U bv;
      #pragma unroll
      for (int j = 0; j < 16; j++) bv.u[j] = Vt[(j>>3)*16 + half*8 + (j&7)][n*16 + l16];
      acc[n] = wmma_bf16(pa, bv, acc[n]);
    }
  }

  #pragma unroll
  for (int r = 0; r < 8; r++){
    int qtok = qt*128 + w*16 + half*8 + r;
    bool qv = qtok < kvalid;                 // invalid txt queries -> 0
    float inv = qv ? (1.f / lrow[r]) : 0.f;
    #pragma unroll
    for (int n = 0; n < 8; n++){
      out[((size_t)b*L + qtok)*HIDc + h*HD + n*16 + l16] = f2bu(acc[n][r] * inv);
    }
  }
}

// ---------------- host orchestration ----------------
extern "C" void kernel_launch(void* const* d_in, const int* in_sizes, int n_in,
                              void* d_out, int out_size, void* d_ws, size_t ws_size,
                              hipStream_t stream)
{
  (void)in_sizes; (void)n_in; (void)out_size; (void)ws_size;
  const float* img = (const float*)d_in[0];
  const float* txt = (const float*)d_in[1];
  const float* pe  = (const float*)d_in[2];
  const int*   tl  = (const int*)d_in[3];
  const float* i1sh=(const float*)d_in[4],  *i1sc=(const float*)d_in[5],  *i1g=(const float*)d_in[6];
  const float* i2sh=(const float*)d_in[7],  *i2sc=(const float*)d_in[8],  *i2g=(const float*)d_in[9];
  const float* t1sh=(const float*)d_in[10], *t1sc=(const float*)d_in[11], *t1g=(const float*)d_in[12];
  const float* t2sh=(const float*)d_in[13], *t2sc=(const float*)d_in[14], *t2g=(const float*)d_in[15];
  const float* iqkv_w=(const float*)d_in[16], *iqkv_b=(const float*)d_in[17];
  const float* iq_s=(const float*)d_in[18],   *ik_s=(const float*)d_in[19];
  const float* iproj_w=(const float*)d_in[20],*iproj_b=(const float*)d_in[21];
  const float* imlp1_w=(const float*)d_in[22],*imlp1_b=(const float*)d_in[23];
  const float* imlp2_w=(const float*)d_in[24],*imlp2_b=(const float*)d_in[25];
  const float* tqkv_w=(const float*)d_in[26], *tqkv_b=(const float*)d_in[27];
  const float* tq_s=(const float*)d_in[28],   *tk_s=(const float*)d_in[29];
  const float* tproj_w=(const float*)d_in[30],*tproj_b=(const float*)d_in[31];
  const float* tmlp1_w=(const float*)d_in[32],*tmlp1_b=(const float*)d_in[33];
  const float* tmlp2_w=(const float*)d_in[34],*tmlp2_b=(const float*)d_in[35];

  // workspace carve (256B aligned)
  char* p = (char*)d_ws;
  auto take = [&](size_t nbytes){ char* r = p; p += (nbytes + 255) & ~(size_t)255; return r; };
  unsigned short* Ximg = (unsigned short*)take(2048ull*3072*2);
  unsigned short* Xtxt = (unsigned short*)take(512ull*3072*2);
  char* bigi = take(2048ull*9216*4);   // QKVimg fp32, later aliased as Himg bf16
  char* bigt = take(512ull*9216*4);    // QKVtxt fp32, later aliased as Htxt bf16
  float* QKVimg = (float*)bigi; unsigned short* Himg = (unsigned short*)bigi;
  float* QKVtxt = (float*)bigt; unsigned short* Htxt = (unsigned short*)bigt;
  unsigned short* Qh = (unsigned short*)take(2ull*24*1280*128*2);
  unsigned short* Kh = (unsigned short*)take(2ull*24*1280*128*2);
  unsigned short* Vh = (unsigned short*)take(2ull*24*1280*128*2);
  unsigned short* AO = (unsigned short*)take(2ull*1280*3072*2);
  float* Rimg = (float*)take(2048ull*3072*4);
  float* Rtxt = (float*)take(512ull*3072*4);
  float* out_img = (float*)d_out;
  float* out_txt = out_img + 2ull*1024*3072;

  // 1) modulated LN (mod1)
  modln_kernel<<<2048, 256, 0, stream>>>(img, i1sh, i1sc, Ximg, 1024);
  modln_kernel<<<512,  256, 0, stream>>>(txt, t1sh, t1sc, Xtxt, 256);

  // 2) QKV projections (N tiles of 64)
  gemm_kernel<0><<<dim3(32, 144), 256, 0, stream>>>(Ximg, iqkv_w, iqkv_b, nullptr, nullptr,
      QKVimg, nullptr, 1024, 9216, 3072, 1024, 0);
  gemm_kernel<0><<<dim3(8, 144), 256, 0, stream>>>(Xtxt, tqkv_w, tqkv_b, nullptr, nullptr,
      QKVtxt, nullptr, 256, 9216, 3072, 256, 0);

  // 3) RMS + RoPE + scatter
  qkv_post_kernel<<<2*24*1280, 128, 0, stream>>>(QKVimg, QKVtxt, pe,
      iq_s, ik_s, tq_s, tk_s, Qh, Kh, Vh);

  // 4) flash attention
  attn_kernel<<<2*24*10, 256, 0, stream>>>(Qh, Kh, Vh, tl, AO);

  // 5) attention projection + gate + residual
  gemm_kernel<2><<<dim3(32, 48), 256, 0, stream>>>(AO, iproj_w, iproj_b, img, i1g,
      Rimg, nullptr, 1024, 3072, 3072, 1280, 0);
  gemm_kernel<2><<<dim3(8, 48), 256, 0, stream>>>(AO, tproj_w, tproj_b, txt, t1g,
      Rtxt, nullptr, 256, 3072, 3072, 1280, 1024);

  // 6) modulated LN (mod2)
  modln_kernel<<<2048, 256, 0, stream>>>(Rimg, i2sh, i2sc, Ximg, 1024);
  modln_kernel<<<512,  256, 0, stream>>>(Rtxt, t2sh, t2sc, Xtxt, 256);

  // 7) MLP1 + GELU -> bf16 hidden (aliases freed QKV scratch)
  gemm_kernel<1><<<dim3(32, 192), 256, 0, stream>>>(Ximg, imlp1_w, imlp1_b, nullptr, nullptr,
      nullptr, Himg, 1024, 12288, 3072, 1024, 0);
  gemm_kernel<1><<<dim3(8, 192), 256, 0, stream>>>(Xtxt, tmlp1_w, tmlp1_b, nullptr, nullptr,
      nullptr, Htxt, 256, 12288, 3072, 256, 0);

  // 8) MLP2 + gate + residual -> final outputs
  gemm_kernel<2><<<dim3(32, 48), 256, 0, stream>>>(Himg, imlp2_w, imlp2_b, Rimg, i2g,
      out_img, nullptr, 1024, 3072, 12288, 1024, 0);
  gemm_kernel<2><<<dim3(8, 48), 256, 0, stream>>>(Htxt, tmlp2_w, tmlp2_b, Rtxt, t2g,
      out_txt, nullptr, 256, 3072, 12288, 256, 0);
}
